// GraphClassifier_14413910245985
// MI455X (gfx1250) — matrix operand order repaired and checked by reading, not compiled, not run on hardware
//
#include <hip/hip_runtime.h>
#include <stdint.h>

typedef __attribute__((ext_vector_type(16))) _Float16 v16h;
typedef __attribute__((ext_vector_type(8)))  float    v8f;
typedef __attribute__((ext_vector_type(4)))  float    f4;

// ---------------------------------------------------------------- utilities
__global__ void zero_kernel(float* __restrict__ p, int n) {
  int i = blockIdx.x * blockDim.x + threadIdx.x;
  if (i < n) p[i] = 0.0f;
}

__global__ void degree_kernel(const int* __restrict__ src, const int* __restrict__ dst,
                              float* __restrict__ odeg, float* __restrict__ ideg, int E) {
  int i = blockIdx.x * blockDim.x + threadIdx.x;
  if (i < E) {
    atomicAdd(&odeg[src[i]], 1.0f);
    atomicAdd(&ideg[dst[i]], 1.0f);
  }
}

__global__ void norm_kernel(float* __restrict__ d, int n) {
  int i = blockIdx.x * blockDim.x + threadIdx.x;
  if (i < n) d[i] = rsqrtf(fmaxf(d[i], 1.0f));
}

// W (64x64 f32, [k][n]) -> Wt (f16, [n][k]) so B-fragments are contiguous
__global__ void transpose_f16_kernel(const float* __restrict__ W, _Float16* __restrict__ Wt) {
  int t = blockIdx.x * blockDim.x + threadIdx.x;
  if (t < 64 * 64) {
    int k = t >> 6, n = t & 63;
    Wt[n * 64 + k] = (_Float16)W[k * 64 + n];
  }
}

__global__ void convert_f16_kernel(const float* __restrict__ s, _Float16* __restrict__ d, int n) {
  int t = blockIdx.x * blockDim.x + threadIdx.x;
  if (t < n) d[t] = (_Float16)s[t];
}

// ------------------------------------------------- WMMA GEMM: m = (h*ns) @ W
// One wave per 16-row tile; K=64 (2 ksteps of 32); N=64 (4 col tiles).
__global__ void __launch_bounds__(256)
gcn_gemm64_kernel(const float* __restrict__ hin, const float* __restrict__ ns,
                  const _Float16* __restrict__ Wt, float* __restrict__ mout, int nrows) {
  int wave = threadIdx.x >> 5;
  int lane = threadIdx.x & 31;
  int tile = blockIdx.x * (blockDim.x >> 5) + wave;
  int rowBase = tile << 4;
  if (rowBase >= nrows) return;
  int half = lane >> 4;       // 0: lanes 0-15, 1: lanes 16-31
  int r = lane & 15;
  int arow = rowBase + r;
  float scale = ns[arow];

  // A-fragment, contiguous K order (matches B):
  // lane<16: K = 0..15 +ks*32 ; lane>=16: K = 16..31 +ks*32
  v16h a[2];
#pragma unroll
  for (int ks = 0; ks < 2; ks++) {
    const float* ap = hin + (size_t)arow * 64 + ks * 32 + half * 16;
    f4 c0 = *(const f4*)(ap);
    f4 c1 = *(const f4*)(ap + 4);
    f4 c2 = *(const f4*)(ap + 8);
    f4 c3 = *(const f4*)(ap + 12);
#pragma unroll
    for (int j = 0; j < 4; j++) {
      a[ks][j]      = (_Float16)(c0[j] * scale);
      a[ks][4 + j]  = (_Float16)(c1[j] * scale);
      a[ks][8 + j]  = (_Float16)(c2[j] * scale);
      a[ks][12 + j] = (_Float16)(c3[j] * scale);
    }
  }

#pragma unroll
  for (int nt = 0; nt < 4; nt++) {
    int col = nt * 16 + r;
    v8f acc = {};
#pragma unroll
    for (int ks = 0; ks < 2; ks++) {
      // B-fragment: lane col, K = ks*32 + (half?16:0) .. +15 contiguous in Wt[n][k]
      v16h b = *(const v16h*)(Wt + col * 64 + ks * 32 + half * 16);
      acc = __builtin_amdgcn_wmma_f32_16x16x32_f16(false, a[ks], false, b,
                                                   (short)0, acc, false, false);
    }
    // D layout: VGPR rr -> M = rr + half*8, N = lane&15
    float* op = mout + (size_t)(rowBase + half * 8) * 64 + col;
#pragma unroll
    for (int rr = 0; rr < 8; rr++) op[rr * 64] = acc[rr];
  }
}

// --------------------------------------------------- edge scatter: agg[dst]+=m[src]
__global__ void scatter_kernel(const float* __restrict__ m, const int* __restrict__ src,
                               const int* __restrict__ dst, float* __restrict__ agg, int E) {
  int tid = blockIdx.x * blockDim.x + threadIdx.x;
  if (tid >= E * 16) return;
  int e = tid >> 4;
  int c = (tid & 15) << 2;
  int s = src[e], d = dst[e];
  f4 v = *(const f4*)(m + (size_t)s * 64 + c);
  float* ap = agg + (size_t)d * 64 + c;
  atomicAdd(ap + 0, v[0]);
  atomicAdd(ap + 1, v[1]);
  atomicAdd(ap + 2, v[2]);
  atomicAdd(ap + 3, v[3]);
}

// h = relu(agg * norm_d + b)
__global__ void finalize_kernel(const float* __restrict__ agg, const float* __restrict__ nd,
                                const float* __restrict__ bias, float* __restrict__ hout, int n) {
  int tid = blockIdx.x * blockDim.x + threadIdx.x;
  if (tid >= n * 16) return;
  int node = tid >> 4;
  int c = (tid & 15) << 2;
  float s = nd[node];
  f4 v = *(const f4*)(agg + (size_t)node * 64 + c);
#pragma unroll
  for (int j = 0; j < 4; j++) {
    float x = v[j] * s + bias[c + j];
    v[j] = x > 0.0f ? x : 0.0f;
  }
  *(f4*)(hout + (size_t)node * 64 + c) = v;
}

// ----------------------------------------- per-node 64-wide bitonic sort (LDS)
__global__ void sort_nodes_kernel(const float* __restrict__ h, float* __restrict__ hs,
                                  float* __restrict__ key, int n) {
  __shared__ float sm[8][64];
  int g = threadIdx.x >> 5;
  int lane = threadIdx.x & 31;
  int node = blockIdx.x * 8 + g;
  bool active = node < n;
  int rn = active ? node : 0;
  sm[g][lane]      = h[(size_t)rn * 64 + lane];
  sm[g][lane + 32] = h[(size_t)rn * 64 + 32 + lane];
  for (int ksz = 2; ksz <= 64; ksz <<= 1) {
    for (int j = ksz >> 1; j > 0; j >>= 1) {
      __syncthreads();
#pragma unroll
      for (int rep = 0; rep < 2; rep++) {
        int i = lane + rep * 32;
        int l = i ^ j;
        if (l > i) {
          float a = sm[g][i], b = sm[g][l];
          bool up = ((i & ksz) == 0);
          if (up ? (a > b) : (a < b)) { sm[g][i] = b; sm[g][l] = a; }
        }
      }
    }
  }
  __syncthreads();
  if (active) {
    hs[(size_t)node * 64 + lane]      = sm[g][lane];
    hs[(size_t)node * 64 + 32 + lane] = sm[g][lane + 32];
    if (lane == 0) key[node] = sm[g][63];  // max (ascending sort)
  }
}

// ----------------------------- stable top-3 per graph (graph_ids are sorted)
__global__ void select_topk_kernel(const float* __restrict__ hs, const float* __restrict__ key,
                                   const int* __restrict__ gid, float* __restrict__ p,
                                   int n, int B) {
  int g = blockIdx.x * blockDim.x + threadIdx.x;
  if (g >= B) return;
  int lo = 0, hi = n;
  while (lo < hi) { int mid = (lo + hi) >> 1; if (gid[mid] < g) lo = mid + 1; else hi = mid; }
  int st = lo;
  lo = st; hi = n;
  while (lo < hi) { int mid = (lo + hi) >> 1; if (gid[mid] < g + 1) lo = mid + 1; else hi = mid; }
  int en = lo;
  float k0 = -1e30f, k1 = -1e30f, k2 = -1e30f;
  int i0 = -1, i1 = -1, i2 = -1;
  for (int i = st; i < en; i++) {
    float kv = key[i];
    if (kv > k0)      { k2 = k1; i2 = i1; k1 = k0; i1 = i0; k0 = kv; i0 = i; }
    else if (kv > k1) { k2 = k1; i2 = i1; k1 = kv; i1 = i; }
    else if (kv > k2) { k2 = kv; i2 = i; }
  }
  int idxs[3] = { i0, i1, i2 };
  for (int rr = 0; rr < 3; rr++) {
    float* op = p + ((size_t)g * 3 + rr) * 64;
    if (idxs[rr] >= 0) {
      const float* ip = hs + (size_t)idxs[rr] * 64;
      for (int c = 0; c < 64; c++) op[c] = ip[c];
    } else {
      for (int c = 0; c < 64; c++) op[c] = 0.0f;
    }
  }
}

// --------------------- WMMA conv: y = relu(P(512x192) @ cwt^T(192x64) + cb)
__global__ void __launch_bounds__(256)
conv_wmma_kernel(const float* __restrict__ P, const _Float16* __restrict__ cwt,
                 const float* __restrict__ cb, float* __restrict__ Y, int nrows) {
  int wave = threadIdx.x >> 5;
  int lane = threadIdx.x & 31;
  int tile = blockIdx.x * (blockDim.x >> 5) + wave;
  int rowBase = tile << 4;
  if (rowBase >= nrows) return;
  int half = lane >> 4;
  int r = lane & 15;
  int arow = rowBase + r;
  v16h a[6];
#pragma unroll
  for (int ks = 0; ks < 6; ks++) {
    const float* ap = P + (size_t)arow * 192 + ks * 32 + half * 16;
    f4 c0 = *(const f4*)(ap);
    f4 c1 = *(const f4*)(ap + 4);
    f4 c2 = *(const f4*)(ap + 8);
    f4 c3 = *(const f4*)(ap + 12);
#pragma unroll
    for (int j = 0; j < 4; j++) {
      a[ks][j]      = (_Float16)c0[j];
      a[ks][4 + j]  = (_Float16)c1[j];
      a[ks][8 + j]  = (_Float16)c2[j];
      a[ks][12 + j] = (_Float16)c3[j];
    }
  }
#pragma unroll
  for (int nt = 0; nt < 4; nt++) {
    int col = nt * 16 + r;
    v8f acc = {};
#pragma unroll
    for (int ks = 0; ks < 6; ks++) {
      v16h b = *(const v16h*)(cwt + col * 192 + ks * 32 + half * 16);
      acc = __builtin_amdgcn_wmma_f32_16x16x32_f16(false, a[ks], false, b,
                                                   (short)0, acc, false, false);
    }
    float bias = cb[col];
    float* op = Y + (size_t)(rowBase + half * 8) * 64 + col;
#pragma unroll
    for (int rr = 0; rr < 8; rr++) {
      float v = acc[rr] + bias;
      op[rr * 64] = v > 0.0f ? v : 0.0f;
    }
  }
}

// ------------------------------------------------ head: out = y @ Wc + bc
__global__ void head_kernel(const float* __restrict__ y, const float* __restrict__ Wc,
                            const float* __restrict__ bc, float* __restrict__ out, int B) {
  int b = blockIdx.x;
  int t = threadIdx.x;
  if (b >= B || t >= 10) return;
  float acc = bc[t];
  for (int c = 0; c < 64; c++) acc += y[(size_t)b * 64 + c] * Wc[c * 10 + t];
  out[(size_t)b * 10 + t] = acc;
}

// ---------------------------------------------------------------- launcher
static inline int cdiv(long a, long b) { return (int)((a + b - 1) / b); }
static inline char* alignup(char* p, size_t a) {
  return (char*)(((uintptr_t)p + a - 1) & ~(uintptr_t)(a - 1));
}

extern "C" void kernel_launch(void* const* d_in, const int* in_sizes, int n_in,
                              void* d_out, int out_size, void* d_ws, size_t ws_size,
                              hipStream_t stream) {
  const float* features = (const float*)d_in[0];
  const int*   esrc     = (const int*)d_in[1];
  const int*   edst     = (const int*)d_in[2];
  const int*   gid      = (const int*)d_in[3];
  const float* W1       = (const float*)d_in[5];
  const float* b1       = (const float*)d_in[6];
  const float* W2       = (const float*)d_in[7];
  const float* b2       = (const float*)d_in[8];
  const float* cw       = (const float*)d_in[9];
  const float* cb       = (const float*)d_in[10];
  const float* Wc       = (const float*)d_in[11];
  const float* bc       = (const float*)d_in[12];
  (void)n_in; (void)ws_size;

  int N = in_sizes[0] / 64;
  int E = in_sizes[1];
  int B = out_size / 10;

  char* w = (char*)d_ws;
  float* bufA = (float*)w; w += (size_t)N * 64 * 4;
  float* bufB = (float*)w; w += (size_t)N * 64 * 4;
  float* odeg = (float*)w; w += (size_t)N * 4;   // ideg contiguous after odeg
  float* ideg = (float*)w; w += (size_t)N * 4;
  float* keyb = (float*)w; w += (size_t)N * 4;
  float* pbuf = (float*)w; w += (size_t)B * 192 * 4;
  float* ybuf = (float*)w; w += (size_t)B * 64 * 4;
  w = alignup(w, 64);
  _Float16* Wt1 = (_Float16*)w; w += 64 * 64 * 2;
  _Float16* Wt2 = (_Float16*)w; w += 64 * 64 * 2;
  _Float16* cwt = (_Float16*)w; w += 64 * 192 * 2;

  const int TB = 256;
  int tiles = cdiv(N, 16);

  // degrees + norms (shared by both layers)
  zero_kernel<<<cdiv(2L * N, TB), TB, 0, stream>>>(odeg, 2 * N);
  degree_kernel<<<cdiv(E, TB), TB, 0, stream>>>(esrc, edst, odeg, ideg, E);
  norm_kernel<<<cdiv(2L * N, TB), TB, 0, stream>>>(odeg, 2 * N);

  // f16 weight prep
  transpose_f16_kernel<<<cdiv(4096, TB), TB, 0, stream>>>(W1, Wt1);
  transpose_f16_kernel<<<cdiv(4096, TB), TB, 0, stream>>>(W2, Wt2);
  convert_f16_kernel<<<cdiv(64 * 192, TB), TB, 0, stream>>>(cw, cwt, 64 * 192);

  // layer 1: m1 -> bufA, agg1 -> bufB, h1 -> bufA
  gcn_gemm64_kernel<<<cdiv(tiles, 5), 160, 0, stream>>>(features, odeg, Wt1, bufA, N);
  zero_kernel<<<cdiv((long)N * 64, TB), TB, 0, stream>>>(bufB, N * 64);
  scatter_kernel<<<cdiv((long)E * 16, TB), TB, 0, stream>>>(bufA, esrc, edst, bufB, E);
  finalize_kernel<<<cdiv((long)N * 16, TB), TB, 0, stream>>>(bufB, ideg, b1, bufA, N);

  // layer 2: m2 -> bufB, agg2 -> bufA, h2 -> bufB
  gcn_gemm64_kernel<<<cdiv(tiles, 5), 160, 0, stream>>>(bufA, odeg, Wt2, bufB, N);
  zero_kernel<<<cdiv((long)N * 64, TB), TB, 0, stream>>>(bufA, N * 64);
  scatter_kernel<<<cdiv((long)E * 16, TB), TB, 0, stream>>>(bufB, esrc, edst, bufA, E);
  finalize_kernel<<<cdiv((long)N * 16, TB), TB, 0, stream>>>(bufA, ideg, b2, bufB, N);

  // sort-pool: hs -> bufA, key per node; then stable top-3 per graph -> pbuf
  sort_nodes_kernel<<<cdiv(N, 8), 256, 0, stream>>>(bufB, bufA, keyb, N);
  select_topk_kernel<<<cdiv(B, TB), TB, 0, stream>>>(bufA, keyb, gid, pbuf, N, B);

  // conv (WMMA, bias+relu fused) then classifier head
  conv_wmma_kernel<<<cdiv(cdiv(B, 16), 8), 256, 0, stream>>>(pbuf, cwt, cb, ybuf, B);
  head_kernel<<<B, 32, 0, stream>>>(ybuf, Wc, bc, (float*)d_out, B);
}
